// PairStack_11605001634122
// MI455X (gfx1250) — matrix operand order repaired and checked by reading, not compiled, not run on hardware
//
#include <hip/hip_runtime.h>
#include <cstdint>

// ---------------------------------------------------------------------------
// Types / helpers
// ---------------------------------------------------------------------------
typedef __bf16 bf16_t;
typedef __attribute__((ext_vector_type(16))) __bf16 v16bf;
typedef __attribute__((ext_vector_type(8)))  float  v8f;

#define DEVINL static __device__ __forceinline__

DEVINL unsigned f2bfbits(float f) {
  union { float f; unsigned u; } v; v.f = f;
  return (v.u + 0x7FFFu + ((v.u >> 16) & 1u)) >> 16;   // RNE
}
DEVINL bf16_t f2bf(float f) {
  union { unsigned short s; bf16_t b; } o; o.s = (unsigned short)f2bfbits(f);
  return o.b;
}
DEVINL float bf2f(bf16_t b) {
  union { bf16_t b; unsigned short s; } i; i.b = b;
  union { unsigned u; float f; } o; o.u = ((unsigned)i.s) << 16;
  return o.f;
}
DEVINL float sigf(float x) { return 1.0f / (1.0f + __expf(-x)); }

DEVINL float hmax16(float v) {          // reduce across a 16-lane half of wave32
#pragma unroll
  for (int off = 1; off < 16; off <<= 1) v = fmaxf(v, __shfl_xor(v, off, 32));
  return v;
}
DEVINL float hsum16(float v) {
#pragma unroll
  for (int off = 1; off < 16; off <<= 1) v += __shfl_xor(v, off, 32);
  return v;
}
DEVINL float hsum32(float v) {
#pragma unroll
  for (int off = 1; off < 32; off <<= 1) v += __shfl_xor(v, off, 32);
  return v;
}

// ---------------------------------------------------------------------------
// Fragment-contiguous swizzle: every 16x32 A-tile / 32x16 B-tile is stored as
// [lane][16] bf16 (512 elems), so a fragment load = one per-lane 32B read.
//   A-tile element (m, k): lane = ((k&15)>>3)*16 + m , e = (k>>4)*8 + (k&7)
//   B-tile element (k, n): lane = ((k&15)>>3)*16 + n , e = (k>>4)*8 + (k&7)
// ---------------------------------------------------------------------------
DEVINL int a_sw_idx(int m, int k) {
  return (((k & 15) >> 3) * 16 + m) * 16 + (k >> 4) * 8 + (k & 7);
}
DEVINL int b_sw_idx(int k, int n) {
  return ((((k & 15) >> 3) * 16 + n) << 4) + (k >> 4) * 8 + (k & 7);
}
DEVINL v16bf load_frag(const bf16_t* tileBase) {
  return *(const v16bf*)(tileBase + (threadIdx.x & 31) * 16);
}
DEVINL v8f wmma_bf16f32(v16bf a, v16bf b, v8f c) {
  return __builtin_amdgcn_wmma_f32_16x16x32_bf16(false, a, false, b, (short)0, c,
                                                 false, false);
}

// ---- async global -> LDS copies (ASYNCcnt-tracked), CDNA5 ----
DEVINL void async_b128(bf16_t* lds_dst, const bf16_t* gsrc) {
  const unsigned lds_off = (unsigned)(uintptr_t)lds_dst;  // low 32b = LDS addr
  asm volatile("global_load_async_to_lds_b128 %0, %1, off"
               :: "v"(lds_off), "v"(gsrc) : "memory");
}
DEVINL void wait_async0() { asm volatile("s_wait_asynccnt 0x0" ::: "memory"); }
DEVINL void wait_async3() { asm volatile("s_wait_asynccnt 0x3" ::: "memory"); }

// ---------------------------------------------------------------------------
// Weight pre-pack: f32 [K][N] -> bf16 B-swizzled [kchunk][ntile][512]
// ---------------------------------------------------------------------------
__global__ __launch_bounds__(256)
void pack_w(const float* __restrict__ W, bf16_t* __restrict__ Wsw, int K, int N) {
  const int t = blockIdx.x * 256 + threadIdx.x;
  if (t >= (K >> 1) * N) return;
  const int n = t % N, kp = t / N, k0 = kp * 2;
  const unsigned lo = f2bfbits(W[(size_t)k0 * N + n]);
  const unsigned hi = f2bfbits(W[(size_t)(k0 + 1) * N + n]);
  const size_t dst = (size_t)(k0 >> 5) * 32 * N + (size_t)(n >> 4) * 512 +
                     b_sw_idx(k0 & 31, n & 15);          // even; +1 = (k0+1,n)
  *(unsigned*)&Wsw[dst] = lo | (hi << 16);
}

// ---------------------------------------------------------------------------
// LayerNorm rows (C=128): one wave per row, bf16 output
// ---------------------------------------------------------------------------
__global__ __launch_bounds__(256)
void ln_rows(const float* __restrict__ x, const float* __restrict__ gam,
             const float* __restrict__ bet, bf16_t* __restrict__ out, int rows) {
  const int row  = blockIdx.x * 8 + (threadIdx.x >> 5);
  const int lane = threadIdx.x & 31;
  if (row >= rows) return;
  float v[4]; float s = 0.f, s2 = 0.f;
#pragma unroll
  for (int i = 0; i < 4; ++i) {
    v[i] = x[(size_t)row * 128 + i * 32 + lane];
    s += v[i]; s2 += v[i] * v[i];
  }
#pragma unroll
  for (int off = 1; off < 32; off <<= 1) {
    s  += __shfl_xor(s,  off, 32);
    s2 += __shfl_xor(s2, off, 32);
  }
  const float mean = s * (1.f / 128.f);
  const float var  = s2 * (1.f / 128.f) - mean * mean;
  const float inv  = rsqrtf(var + 1e-5f);
#pragma unroll
  for (int i = 0; i < 4; ++i) {
    const int c = i * 32 + lane;
    out[(size_t)row * 128 + c] = f2bf((v[i] - mean) * inv * gam[c] + bet[c]);
  }
}

// ---------------------------------------------------------------------------
// Generic WMMA GEMM:  Y[M x Ntot] = epilogue( X(bf16) @ Wsw(bf16 swizzled) )
// WG = 256 threads = 8 waves (2 M x 4 N), tile 64 x 128, K chunked by 32.
// MODE 1 sigmoid bf16 [p][Ntot]      MODE 2 relu bf16 [p][Ntot]
// MODE 3 dual-gate -> tri A-operand swizzled global   (TRANS swaps p1/p2)
// MODE 5 dual-gate -> tri B-operand swizzled global
// MODE 4 residual: resid[r][c] += (acc + b[c]) * optional gate[r][c]
// MODE 6/7/8: attention q/k/v swizzled stores (TRANS = ending orientation)
// ---------------------------------------------------------------------------
template<int MODE, bool TRANS>
__global__ __launch_bounds__(256)
void gemm128(const bf16_t* __restrict__ X, int ldx,
             const bf16_t* __restrict__ Wb, const float* __restrict__ bias,
             const bf16_t* __restrict__ W2b, const float* __restrict__ bias2,
             const bf16_t* __restrict__ gate,
             bf16_t* __restrict__ outb,
             float* __restrict__ resid,
             int K, int Ntot) {
  constexpr bool DUAL = (MODE == 3 || MODE == 5);
  __shared__ alignas(32) bf16_t Xs[4 * 512];
  __shared__ alignas(32) bf16_t Ws[8 * 512];
  __shared__ alignas(32) bf16_t Ws2[DUAL ? 8 * 512 : 16];

  const int tid = threadIdx.x, wid = tid >> 5;
  const int wm = wid >> 2, wn = wid & 3;
  const int lane = tid & 31, half = lane >> 4, nn = lane & 15;
  const int row0 = blockIdx.x * 64;
  const int col0 = blockIdx.y * 128;

  v8f acc[2][2]  = {};
  v8f acc2[2][2] = {};

  for (int kc = 0; kc < K; kc += 32) {
    __syncthreads();
    {  // stage X via async LDS-DMA into A-swizzle (one b128 per thread)
      const int r64 = tid >> 2, g = (tid >> 1) & 1, hf = tid & 1;
      async_b128(&Xs[(r64 >> 4) * 512 + (hf * 16 + (r64 & 15)) * 16 + g * 8],
                 &X[(size_t)(row0 + r64) * ldx + kc + g * 16 + hf * 8]);
    }
    {  // stage pre-swizzled weights (pure async copies)
      const size_t wb = (size_t)(kc >> 5) * 32 * Ntot + (size_t)(col0 >> 4) * 512;
      async_b128(&Ws[tid * 8],        &Wb[wb + tid * 8]);
      async_b128(&Ws[2048 + tid * 8], &Wb[wb + 2048 + tid * 8]);
      if (DUAL) {
        async_b128(&Ws2[tid * 8],        &W2b[wb + tid * 8]);
        async_b128(&Ws2[2048 + tid * 8], &W2b[wb + 2048 + tid * 8]);
      }
    }
    wait_async0();
    __syncthreads();

    v16bf af[2];
#pragma unroll
    for (int qg = 0; qg < 2; ++qg)
      af[qg] = load_frag(&Xs[(wm * 2 + qg) * 512]);
#pragma unroll
    for (int t = 0; t < 2; ++t) {
      v16bf bfb = load_frag(&Ws[(wn * 2 + t) * 512]);
#pragma unroll
      for (int qg = 0; qg < 2; ++qg)
        acc[qg][t] = wmma_bf16f32(af[qg], bfb, acc[qg][t]);
      if (DUAL) {
        v16bf bf2 = load_frag(&Ws2[(wn * 2 + t) * 512]);
#pragma unroll
        for (int qg = 0; qg < 2; ++qg)
          acc2[qg][t] = wmma_bf16f32(af[qg], bf2, acc2[qg][t]);
      }
    }
  }

#pragma unroll
  for (int qg = 0; qg < 2; ++qg)
#pragma unroll
    for (int t = 0; t < 2; ++t)
#pragma unroll
      for (int v = 0; v < 8; ++v) {
        const int r = row0 + wm * 32 + qg * 16 + half * 8 + v;
        const int c = col0 + wn * 32 + t * 16 + nn;
        const int p1 = r >> 8, p2 = r & 255;
        const int first  = TRANS ? p2 : p1;
        const int second = TRANS ? p1 : p2;
        float y = acc[qg][t][v];
        if (MODE == 1) {
          outb[(size_t)r * Ntot + c] = f2bf(sigf(y + bias[c]));
        } else if (MODE == 2) {
          outb[(size_t)r * Ntot + c] = f2bf(fmaxf(y + bias[c], 0.f));
        } else if (MODE == 3 || MODE == 5) {
          const float gv = sigf(y + bias[c]);
          const float pv = acc2[qg][t][v] + bias2[c];
          size_t addr;
          if (MODE == 3)     // A operand: row=first, k=second
            addr = (size_t)c * 65536 + (size_t)(second >> 5) * 8192 +
                   (size_t)(first >> 4) * 512 + a_sw_idx(first & 15, second & 31);
          else               // B operand: k=first, col=second
            addr = (size_t)c * 65536 + (size_t)(first >> 5) * 8192 +
                   (size_t)(second >> 4) * 512 + b_sw_idx(first & 31, second & 15);
          outb[addr] = f2bf(gv * pv);
        } else if (MODE == 4) {
          y += bias[c];
          if (gate) y *= bf2f(gate[(size_t)r * 128 + c]);
          resid[(size_t)r * 128 + c] += y;
        } else {  // 6/7/8: attention operands, r_=first (fixed), u=second
          const int hh = c >> 5, d = c & 31;
          const size_t base = (size_t)hh * 2097152 + (size_t)first * 8192;
          size_t addr;
          if (MODE == 6)        // Q: A-tiles over queries
            addr = base + (size_t)(second >> 4) * 512 + a_sw_idx(second & 15, d);
          else if (MODE == 7)   // K: B-tiles [d][key]
            addr = base + (size_t)(second >> 5) * 1024 +
                   (size_t)((second & 31) >> 4) * 512 + b_sw_idx(d, second & 15);
          else                  // V: B-tiles [key][d]
            addr = base + (size_t)(second >> 5) * 1024 +
                   (size_t)(d >> 4) * 512 + b_sw_idx(second & 31, d & 15);
          outb[addr] = f2bf(y + bias[c]);
        }
      }
}

// ---------------------------------------------------------------------------
// Triangle einsum: per channel c: T_c = A_c @ B_c, operands pre-swizzled.
// Double-buffered async global->LDS pipeline: issue chunk k+1, wait ASYNCcnt<=3
// (in-order completion => chunk k landed), compute chunk k.
// grid (c=128, itile=4, jtile=2)
// ---------------------------------------------------------------------------
__global__ __launch_bounds__(256)
void tri_gemm(const bf16_t* __restrict__ A, const bf16_t* __restrict__ Bm,
              float* __restrict__ T) {
  __shared__ alignas(32) bf16_t As[2][2048];
  __shared__ alignas(32) bf16_t Bs[2][4096];
  const int c  = blockIdx.x;
  const int i0 = blockIdx.y * 64;
  const int j0 = blockIdx.z * 128;
  const int tid = threadIdx.x, wid = tid >> 5;
  const int wm = wid >> 2, wn = wid & 3;
  const int lane = tid & 31, half = lane >> 4, nn = lane & 15;
  const bf16_t* GA = A  + (size_t)c * 65536 + (size_t)(i0 >> 4) * 512;
  const bf16_t* GB = Bm + (size_t)c * 65536 + (size_t)(j0 >> 4) * 512;

  // prologue: issue chunk 0 into buffer 0
  async_b128(&As[0][tid * 8],        GA + tid * 8);
  async_b128(&Bs[0][tid * 8],        GB + tid * 8);
  async_b128(&Bs[0][2048 + tid * 8], GB + 2048 + tid * 8);

  v8f acc[2][2] = {};
  for (int ch = 0; ch < 8; ++ch) {
    const int cur = ch & 1;
    if (ch + 1 < 8) {   // issue next chunk into the other buffer
      const size_t ko = (size_t)(ch + 1) * 8192;
      async_b128(&As[cur ^ 1][tid * 8],        GA + ko + tid * 8);
      async_b128(&Bs[cur ^ 1][tid * 8],        GB + ko + tid * 8);
      async_b128(&Bs[cur ^ 1][2048 + tid * 8], GB + ko + 2048 + tid * 8);
      wait_async3();    // current chunk's 3 copies (older) have completed
    } else {
      wait_async0();
    }
    __syncthreads();

    v16bf af[2];
#pragma unroll
    for (int qg = 0; qg < 2; ++qg)
      af[qg] = load_frag(&As[cur][(wm * 2 + qg) * 512]);
#pragma unroll
    for (int t = 0; t < 2; ++t) {
      v16bf bfb = load_frag(&Bs[cur][(wn * 2 + t) * 512]);
#pragma unroll
      for (int qg = 0; qg < 2; ++qg)
        acc[qg][t] = wmma_bf16f32(af[qg], bfb, acc[qg][t]);
    }
    __syncthreads();    // all waves done reading buf[cur] before it is refilled
  }
#pragma unroll
  for (int qg = 0; qg < 2; ++qg)
#pragma unroll
    for (int t = 0; t < 2; ++t)
#pragma unroll
      for (int v = 0; v < 8; ++v) {
        const int i = i0 + wm * 32 + qg * 16 + half * 8 + v;
        const int j = j0 + wn * 32 + t * 16 + nn;
        T[((size_t)i * 256 + j) * 128 + c] = acc[qg][t][v];
      }
}

// ---------------------------------------------------------------------------
// Bias projection 128 -> 4 heads: one wave per position.
// trans=1 writes B0[h][b][a] so attention always reads bias[h][query][key].
// ---------------------------------------------------------------------------
__global__ __launch_bounds__(256)
void bias_proj(const bf16_t* __restrict__ zn, const float* __restrict__ bw,
               float* __restrict__ b0, int trans) {
  const int p    = blockIdx.x * 8 + (threadIdx.x >> 5);
  const int lane = threadIdx.x & 31;
  float a[4] = {0.f, 0.f, 0.f, 0.f};
#pragma unroll
  for (int i = 0; i < 4; ++i) {
    const int c = i * 32 + lane;
    const float xv = bf2f(zn[(size_t)p * 128 + c]);
#pragma unroll
    for (int h = 0; h < 4; ++h) a[h] += xv * bw[c * 4 + h];
  }
#pragma unroll
  for (int h = 0; h < 4; ++h) a[h] = hsum32(a[h]);
  const int q = trans ? ((p & 255) * 256 + (p >> 8)) : p;
  if (lane == 0)
#pragma unroll
    for (int h = 0; h < 4; ++h) b0[h * 65536 + q] = a[h];
}

// ---------------------------------------------------------------------------
// Triangle attention (flash-style). WG per (r, h); 8 waves x 32 queries.
// q/k/v pre-swizzled per (h, r): Q A-tiles; K/V B-tiles, 1024 elems per chunk
// staged via async LDS-DMA. bias(u,w) = B0[h*65536 + u*256 + w] (coalesced).
// G/O remain [pos][128]; pos = r*sR + u*sX.
// ---------------------------------------------------------------------------
__global__ __launch_bounds__(256)
void tri_attn(const bf16_t* __restrict__ Q, const bf16_t* __restrict__ Kb,
              const bf16_t* __restrict__ Vb, const bf16_t* __restrict__ G,
              const float* __restrict__ B0, bf16_t* __restrict__ O,
              int sR, int sX) {
  __shared__ alignas(32) bf16_t Kt[2 * 512];
  __shared__ alignas(32) bf16_t Vs[2 * 512];
  __shared__ alignas(32) bf16_t Ps[8][2 * 512];
  const int r = blockIdx.x, h = blockIdx.y;
  const int tid = threadIdx.x, wid = tid >> 5;
  const int lane = tid & 31, half = lane >> 4, nn = lane & 15;
  const int u0 = wid * 32;
  const float scale = 0.17677669529663687f;  // 1/sqrt(32)
  const size_t hbase = (size_t)h * 2097152 + (size_t)r * 8192;

  v16bf qa[2];
#pragma unroll
  for (int qg = 0; qg < 2; ++qg)   // direct per-lane 32B fragment loads
    qa[qg] = *(const v16bf*)&Q[hbase + (size_t)(wid * 2 + qg) * 512 + lane * 16];

  v8f oacc[2][2] = {};
  float mrun[2][8], lrun[2][8];
#pragma unroll
  for (int qg = 0; qg < 2; ++qg)
#pragma unroll
    for (int v = 0; v < 8; ++v) { mrun[qg][v] = -3.0e38f; lrun[qg][v] = 0.f; }

  for (int w0 = 0; w0 < 256; w0 += 32) {
    __syncthreads();
    const size_t cbase = hbase + (size_t)(w0 >> 5) * 1024;
    if (tid < 128)
      async_b128(&Kt[tid * 8], &Kb[cbase + tid * 8]);
    else
      async_b128(&Vs[(tid - 128) * 8], &Vb[cbase + (tid - 128) * 8]);
    wait_async0();
    __syncthreads();

#pragma unroll
    for (int qg = 0; qg < 2; ++qg) {
      v8f s[2];
#pragma unroll
      for (int t = 0; t < 2; ++t) {
        v8f zacc = {};
        s[t] = wmma_bf16f32(qa[qg], load_frag(&Kt[t * 512]), zacc);
      }
      float cm[8];
#pragma unroll
      for (int v = 0; v < 8; ++v) {
        const int u = u0 + qg * 16 + half * 8 + v;
#pragma unroll
        for (int t = 0; t < 2; ++t) {
          const int w = w0 + t * 16 + nn;
          s[t][v] = s[t][v] * scale + B0[(size_t)h * 65536 + (size_t)u * 256 + w];
        }
        cm[v] = hmax16(fmaxf(s[0][v], s[1][v]));
      }
#pragma unroll
      for (int v = 0; v < 8; ++v) {
        const float mn    = fmaxf(mrun[qg][v], cm[v]);
        const float alpha = __expf(mrun[qg][v] - mn);
        mrun[qg][v] = mn;
        float rs = 0.f;
#pragma unroll
        for (int t = 0; t < 2; ++t) {
          const float p = __expf(s[t][v] - mn);
          s[t][v] = p;
          rs += p;
        }
        rs = hsum16(rs);
        lrun[qg][v] = lrun[qg][v] * alpha + rs;
        oacc[qg][0][v] *= alpha;
        oacc[qg][1][v] *= alpha;
      }
      // P (D-layout) -> A-swizzled LDS tile, wave-private
#pragma unroll
      for (int t = 0; t < 2; ++t)
#pragma unroll
        for (int v = 0; v < 8; ++v) {
          const int idx = qg * 512 + ((nn >> 3) * 16 + half * 8 + v) * 16 +
                          t * 8 + (nn & 7);
          Ps[wid][idx] = f2bf(s[t][v]);
        }
    }
#pragma unroll
    for (int qg = 0; qg < 2; ++qg) {
      v16bf pa = load_frag(&Ps[wid][qg * 512]);
#pragma unroll
      for (int dg = 0; dg < 2; ++dg)
        oacc[qg][dg] = wmma_bf16f32(pa, load_frag(&Vs[dg * 512]), oacc[qg][dg]);
    }
  }

#pragma unroll
  for (int qg = 0; qg < 2; ++qg)
#pragma unroll
    for (int dg = 0; dg < 2; ++dg)
#pragma unroll
      for (int v = 0; v < 8; ++v) {
        const int u = u0 + qg * 16 + half * 8 + v;
        const int d = dg * 16 + nn;
        const size_t pos = (size_t)(r * sR + u * sX);
        float ov = oacc[qg][dg][v] / lrun[qg][v];
        ov *= bf2f(G[pos * 128 + h * 32 + d]);
        O[pos * 128 + h * 32 + d] = f2bf(ov);
      }
}

// ---------------------------------------------------------------------------
// Host orchestration
// ---------------------------------------------------------------------------
extern "C" void kernel_launch(void* const* d_in, const int* in_sizes, int n_in,
                              void* d_out, int out_size, void* d_ws, size_t ws_size,
                              hipStream_t stream) {
  (void)in_sizes; (void)n_in; (void)out_size; (void)ws_size;
  const int P = 256 * 256;
  float* z = (float*)d_out;
  hipMemcpyAsync(z, d_in[0], (size_t)P * 128 * sizeof(float),
                 hipMemcpyDeviceToDevice, stream);

  // workspace layout (needs ~178 MB)
  char* ws = (char*)d_ws;
  bf16_t* zn   = (bf16_t*)(ws);                            // 16 MB
  bf16_t* Ab   = (bf16_t*)(ws + ((size_t)16  << 20));      // 16 MB  (tri a / q)
  bf16_t* Bb   = (bf16_t*)(ws + ((size_t)32  << 20));      // 16 MB  (tri b / k)
  bf16_t* Gb   = (bf16_t*)(ws + ((size_t)48  << 20));      // 16 MB  (gates)
  float*  Tb   = (float*) (ws + ((size_t)64  << 20));      // 32 MB  (tri out f32)
  bf16_t* Vbuf = (bf16_t*)(ws + ((size_t)64  << 20));      // overlays Tb (attn v)
  float*  B0   = (float*) (ws + ((size_t)80  << 20));      // 1 MB   (attn bias)
  bf16_t* TN   = (bf16_t*)(ws + ((size_t)96  << 20));      // 16 MB  (tn / attn o)
  bf16_t* Hb   = (bf16_t*)(ws + ((size_t)112 << 20));      // 64 MB  (transition)
  bf16_t* Wp   = (bf16_t*)(ws + ((size_t)176 << 20));      // ~1 MB  packed weights

  const dim3 blk(256);
  const dim3 g_ln(P / 8);
  const dim3 g_gm(P / 64, 1);
  const dim3 g_gm4(P / 64, 4);
  const dim3 g_tri(128, 4, 2);
  const dim3 g_at(256, 4);

  auto F = [&](int i) { return (const float*)d_in[i]; };

  // ---- pre-pack all weight matrices into B-swizzled bf16 (once per call)
  size_t woff = 0;
  auto pack = [&](int idx, int K, int N) {
    bf16_t* dst = Wp + woff;
    woff += (size_t)K * N;
    const int tasks = (K / 2) * N;
    pack_w<<<dim3((tasks + 255) / 256), blk, 0, stream>>>(F(idx), dst, K, N);
    return dst;
  };
  bf16_t *wAg[2], *wAp[2], *wBg[2], *wBp[2], *wG[2], *wZ[2];
  for (int m = 0; m < 2; ++m) {
    const int b = 1 + m * 16;
    wAg[m] = pack(b + 2, 128, 128);  wAp[m] = pack(b + 4, 128, 128);
    wBg[m] = pack(b + 6, 128, 128);  wBp[m] = pack(b + 8, 128, 128);
    wG[m]  = pack(b + 10, 128, 128); wZ[m]  = pack(b + 14, 128, 128);
  }
  bf16_t *wQ[2], *wK[2], *wV[2], *wGt[2], *wO[2];
  for (int m = 0; m < 2; ++m) {
    const int b = 33 + m * 13;
    wQ[m] = pack(b + 3, 128, 128);  wK[m]  = pack(b + 5, 128, 128);
    wV[m] = pack(b + 7, 128, 128);  wGt[m] = pack(b + 9, 128, 128);
    wO[m] = pack(b + 11, 128, 128);
  }
  bf16_t* wL1 = pack(61, 128, 512);
  bf16_t* wL2 = pack(63, 512, 128);

  // ---- triangle multiplication (outgoing m=0, incoming m=1)
  for (int m = 0; m < 2; ++m) {
    const int b = 1 + m * 16;
    ln_rows<<<g_ln, blk, 0, stream>>>(z, F(b + 0), F(b + 1), zn, P);
    if (m == 0) {
      gemm128<3, false><<<g_gm, blk, 0, stream>>>(zn, 128, wAg[m], F(b+3), wAp[m], F(b+5),
                                                  nullptr, Ab, nullptr, 128, 128);
      gemm128<5, true ><<<g_gm, blk, 0, stream>>>(zn, 128, wBg[m], F(b+7), wBp[m], F(b+9),
                                                  nullptr, Bb, nullptr, 128, 128);
    } else {
      gemm128<3, true ><<<g_gm, blk, 0, stream>>>(zn, 128, wAg[m], F(b+3), wAp[m], F(b+5),
                                                  nullptr, Ab, nullptr, 128, 128);
      gemm128<5, false><<<g_gm, blk, 0, stream>>>(zn, 128, wBg[m], F(b+7), wBp[m], F(b+9),
                                                  nullptr, Bb, nullptr, 128, 128);
    }
    gemm128<1, false><<<g_gm, blk, 0, stream>>>(zn, 128, wG[m], F(b+11), nullptr, nullptr,
                                                nullptr, Gb, nullptr, 128, 128);
    tri_gemm<<<g_tri, blk, 0, stream>>>(Ab, Bb, Tb);
    ln_rows<<<g_ln, blk, 0, stream>>>(Tb, F(b + 12), F(b + 13), TN, P);
    gemm128<4, false><<<g_gm, blk, 0, stream>>>(TN, 128, wZ[m], F(b+15), nullptr, nullptr,
                                                Gb, nullptr, z, 128, 128);
  }

  // ---- triangle attention (starting m=0, ending m=1)
  for (int m = 0; m < 2; ++m) {
    const int b = 33 + m * 13;
    ln_rows<<<g_ln, blk, 0, stream>>>(z, F(b + 0), F(b + 1), zn, P);
    if (m == 0) {
      gemm128<6, false><<<g_gm, blk, 0, stream>>>(zn, 128, wQ[m], F(b+4), nullptr, nullptr,
                                                  nullptr, Ab, nullptr, 128, 128);
      gemm128<7, false><<<g_gm, blk, 0, stream>>>(zn, 128, wK[m], F(b+6), nullptr, nullptr,
                                                  nullptr, Bb, nullptr, 128, 128);
      gemm128<8, false><<<g_gm, blk, 0, stream>>>(zn, 128, wV[m], F(b+8), nullptr, nullptr,
                                                  nullptr, Vbuf, nullptr, 128, 128);
    } else {
      gemm128<6, true ><<<g_gm, blk, 0, stream>>>(zn, 128, wQ[m], F(b+4), nullptr, nullptr,
                                                  nullptr, Ab, nullptr, 128, 128);
      gemm128<7, true ><<<g_gm, blk, 0, stream>>>(zn, 128, wK[m], F(b+6), nullptr, nullptr,
                                                  nullptr, Bb, nullptr, 128, 128);
      gemm128<8, true ><<<g_gm, blk, 0, stream>>>(zn, 128, wV[m], F(b+8), nullptr, nullptr,
                                                  nullptr, Vbuf, nullptr, 128, 128);
    }
    gemm128<1, false><<<g_gm, blk, 0, stream>>>(zn, 128, wGt[m], F(b+10), nullptr, nullptr,
                                                nullptr, Gb, nullptr, 128, 128);
    bias_proj<<<dim3(P / 8), blk, 0, stream>>>(zn, F(b + 2), B0, m);
    if (m == 0)
      tri_attn<<<g_at, blk, 0, stream>>>(Ab, Bb, Vbuf, Gb, B0, TN, 256, 1);
    else
      tri_attn<<<g_at, blk, 0, stream>>>(Ab, Bb, Vbuf, Gb, B0, TN, 1, 256);
    gemm128<4, false><<<g_gm, blk, 0, stream>>>(TN, 128, wO[m], F(b+12), nullptr, nullptr,
                                                nullptr, nullptr, z, 128, 128);
  }

  // ---- pair transition (pt base 59)
  ln_rows<<<g_ln, blk, 0, stream>>>(z, F(59), F(60), zn, P);
  gemm128<2, false><<<g_gm4, blk, 0, stream>>>(zn, 128, wL1, F(62), nullptr, nullptr,
                                               nullptr, Hb, nullptr, 128, 512);
  gemm128<4, false><<<g_gm, blk, 0, stream>>>(Hb, 512, wL2, F(64), nullptr, nullptr,
                                              nullptr, nullptr, z, 512, 128);
}
